// hyp_model_1_54013508714677
// MI455X (gfx1250) — matrix operand order repaired and checked by reading, not compiled, not run on hardware
//
#include <hip/hip_runtime.h>
#include <hip/hip_bf16.h>
#include <stdint.h>

typedef long long i64;
typedef __attribute__((ext_vector_type(16))) __bf16 v16bf;
typedef __attribute__((ext_vector_type(8)))  float  v8f;

#define NBATCH 4
#define NPTS   1024
#define DIM    160
#define NCL    10
#define KTOP   128
#define NPAIR  8128   // K*(K-1)/2
#define NHEADS 5
#define DH     32
#define DFF    640
#define NLAY   7
#define OUTD   64

// ---------------------------------------------------------------------------
// helpers
// ---------------------------------------------------------------------------
static __device__ inline __bf16 f2bf(float f) {
    return (__bf16)f;   // native RNE convert (v_cvt_*bf16_f32 on gfx1250)
}

static __device__ inline float waveRedSum(float v) {
    #pragma unroll
    for (int m = 16; m > 0; m >>= 1) v += __shfl_xor(v, m, 32);
    return v;
}
static __device__ inline float waveRedMax(float v) {
    #pragma unroll
    for (int m = 16; m > 0; m >>= 1) v = fmaxf(v, __shfl_xor(v, m, 32));
    return v;
}

// ---------------------------------------------------------------------------
// Generic strided batched GEMM: C = act(A @ B + bias)
// A(m,k) = Ab[m*sAr + k]                (K-contiguous at every call site)
// B(k,n) = Bb[k*sBr + n*sBc]
// C(m,n) = Cb[m*sCr + n]
// Batch z: base offset = (z/nInner)*so + (z%nInner)*si (per matrix).
// One wave computes a 16x32 C strip: one A fragment reused by two
// back-to-back v_wmma_f32_16x16x32_bf16 accumulations per k-step.
// M % 16 == 0, N % 32 == 0, K % 32 == 0 (true for all call sites).
// ---------------------------------------------------------------------------
__global__ __launch_bounds__(256)
void gemm_wmma(const float* __restrict__ A, i64 sAr, i64 soA, i64 siA,
               const float* __restrict__ B, i64 sBr, i64 sBc, i64 soB, i64 siB,
               const float* __restrict__ bias, i64 soBi, i64 siBi,
               float* __restrict__ C, i64 sCr, i64 soC, i64 siC,
               int M, int N, int Kd, int nInner, int relu)
{
    int z  = blockIdx.y;
    int zo = z / nInner, zi = z % nInner;
    const float* Ab = A + zo * soA + zi * siA;
    const float* Bb = B + zo * soB + zi * siB;
    float*       Cb = C + zo * soC + zi * siC;
    const float* Bi = bias ? (bias + zo * soBi + zi * siBi) : nullptr;

    int tilesN = N >> 5;                       // 32-wide strips
    int tilesM = M >> 4;
    int wave = blockIdx.x * (blockDim.x >> 5) + (threadIdx.x >> 5);
    if (wave >= tilesM * tilesN) return;       // wave-uniform: EXEC stays all-ones
    int tm = wave / tilesN, tn = wave % tilesN;

    int lane = threadIdx.x & 31;
    int half = lane >> 4;
    int rc   = lane & 15;

    // per-lane K offsets for the 8 VGPR slots (element pairs are K-contiguous)
    int kaOff[8], kbOff[8];
    #pragma unroll
    for (int v = 0; v < 8; ++v) {
        kaOff[v] = ((v >> 2) << 4) + (half << 3) + ((v & 3) << 1);  // A 16x32 layout
        kbOff[v] = ((v >> 2) << 4) + (half << 3) + ((v & 3) << 1);  // B 32x16 layout (same K order as A)
    }

    const float* Arow = Ab + (i64)(tm * 16 + rc) * sAr;
    i64 bcol0 = (i64)(tn * 32 + rc)      * sBc;
    i64 bcol1 = (i64)(tn * 32 + 16 + rc) * sBc;
    bool contigB = (sBr == 1);                 // K-contiguous B (S = Q K^T case)

    v8f acc0 = {}, acc1 = {};
    for (int k0 = 0; k0 < Kd; k0 += 32) {
        v16bf a, b0, b1;
        #pragma unroll
        for (int v = 0; v < 8; ++v) {
            float2 av = *(const float2*)(Arow + k0 + kaOff[v]);
            a[2 * v]     = f2bf(av.x);
            a[2 * v + 1] = f2bf(av.y);
            int kb = k0 + kbOff[v];
            if (contigB) {
                float2 u0 = *(const float2*)(Bb + kb + bcol0);
                float2 u1 = *(const float2*)(Bb + kb + bcol1);
                b0[2 * v] = f2bf(u0.x); b0[2 * v + 1] = f2bf(u0.y);
                b1[2 * v] = f2bf(u1.x); b1[2 * v + 1] = f2bf(u1.y);
            } else {
                b0[2 * v]     = f2bf(Bb[(i64)(kb)     * sBr + bcol0]);
                b0[2 * v + 1] = f2bf(Bb[(i64)(kb + 1) * sBr + bcol0]);
                b1[2 * v]     = f2bf(Bb[(i64)(kb)     * sBr + bcol1]);
                b1[2 * v + 1] = f2bf(Bb[(i64)(kb + 1) * sBr + bcol1]);
            }
        }
        acc0 = __builtin_amdgcn_wmma_f32_16x16x32_bf16(
            false, a, false, b0, (short)0, acc0, false, false);
        acc1 = __builtin_amdgcn_wmma_f32_16x16x32_bf16(
            false, a, false, b1, (short)0, acc1, false, false);
    }

    // branchless epilogue: bias hoisted per column, relu folded into fmaxf
    int col0 = tn * 32 + rc, col1 = col0 + 16;
    float bv0 = Bi ? Bi[col0] : 0.f;
    float bv1 = Bi ? Bi[col1] : 0.f;
    float lo  = relu ? 0.f : -3.4e38f;
    #pragma unroll
    for (int v = 0; v < 8; ++v) {
        int row = tm * 16 + half * 8 + v;      // C layout: M = v + half*8
        Cb[(i64)row * sCr + col0] = fmaxf(acc0[v] + bv0, lo);
        Cb[(i64)row * sCr + col1] = fmaxf(acc1[v] + bv1, lo);
    }
}

// ---------------------------------------------------------------------------
// per-pixel embedding: h[b,n,d] = sum_c x[b,c,n]*W[c,d] + be[d]
// ---------------------------------------------------------------------------
__global__ void embed_k(const float* __restrict__ x, const float* __restrict__ w,
                        const float* __restrict__ be, float* __restrict__ h)
{
    int i = blockIdx.x * blockDim.x + threadIdx.x;
    if (i >= NBATCH * NPTS * DIM) return;
    int d = i % DIM; int bn = i / DIM;
    int n = bn % NPTS; int b = bn / NPTS;
    float acc = be[d];
    #pragma unroll
    for (int c = 0; c < 3; ++c)
        acc += x[((i64)(b * 3 + c)) * NPTS + n] * w[c * DIM + d];
    h[i] = acc;
}

// sims[b,k,n] = <h[b,n,:], centers[k,:]>
__global__ void sims_k(const float* __restrict__ h, const float* __restrict__ cen,
                       float* __restrict__ sims)
{
    int i = blockIdx.x * blockDim.x + threadIdx.x;
    if (i >= NBATCH * NCL * NPTS) return;
    int n = i % NPTS; int k = (i / NPTS) % NCL; int b = i / (NPTS * NCL);
    const float* hv = h + ((i64)(b * NPTS + n)) * DIM;
    const float* cv = cen + k * DIM;
    float a = 0.f;
    for (int d = 0; d < DIM; ++d) a += hv[d] * cv[d];
    sims[i] = a;
}

// exact top-K by rank (tie-break: lower index first, matches lax.top_k)
__global__ __launch_bounds__(1024)
void topk_k(const float* __restrict__ sims, int* __restrict__ idx)
{
    __shared__ float sv[NPTS];
    int t = threadIdx.x;
    int bk = blockIdx.x;                    // (b*NCL + k)
    const float* row = sims + (i64)bk * NPTS;
    sv[t] = row[t];
    __syncthreads();
    float v = sv[t];
    int rank = 0;
    for (int j = 0; j < NPTS; ++j) {
        float u = sv[j];
        rank += (u > v) || (u == v && j < t);
    }
    if (rank < KTOP) idx[bk * KTOP + rank] = t;
}

// triu pair table (i<j) for K=128
__global__ void pairs_k(int* __restrict__ pi, int* __restrict__ pj)
{
    int p = blockIdx.x * blockDim.x + threadIdx.x;
    if (p >= NPAIR) return;
    int i = 0, rem = p;
    while (rem >= (KTOP - 1 - i)) { rem -= (KTOP - 1 - i); ++i; }
    pi[p] = i;
    pj[p] = i + 1 + rem;
}

__global__ void deginit_k(float* __restrict__ deg)
{
    int i = blockIdx.x * blockDim.x + threadIdx.x;
    if (i < NBATCH * NPTS) deg[i] = 1.0f;     // self-loop ones
}

__global__ void degadd_k(const int* __restrict__ topk, const int* __restrict__ pi,
                         const int* __restrict__ pj, float* __restrict__ deg)
{
    int t = blockIdx.x * blockDim.x + threadIdx.x;
    const int tot = NBATCH * NCL * NPAIR;
    if (t >= tot) return;
    int p = t % NPAIR; int c = (t / NPAIR) % NCL; int b = t / (NPAIR * NCL);
    const int* row = topk + (b * NCL + c) * KTOP;
    int s = row[pi[p]], d = row[pj[p]];
    atomicAdd(&deg[b * NPTS + s], 1.0f);
    atomicAdd(&deg[b * NPTS + d], 1.0f);
}

__global__ void zero_k(float* __restrict__ p, i64 n)
{
    i64 i = (i64)blockIdx.x * blockDim.x + threadIdx.x;
    if (i < n) p[i] = 0.f;
}

// symmetric-normalized scatter: both directions of each pair
__global__ void gcnagg_k(const int* __restrict__ topk, const int* __restrict__ pi,
                         const int* __restrict__ pj, const float* __restrict__ deg,
                         const float* __restrict__ hw, float* __restrict__ agg)
{
    i64 t = (i64)blockIdx.x * blockDim.x + threadIdx.x;
    const i64 tot = (i64)NBATCH * NCL * NPAIR * DIM;
    if (t >= tot) return;
    int d = (int)(t % DIM);
    i64 e = t / DIM;
    int p = (int)(e % NPAIR); int c = (int)((e / NPAIR) % NCL); int b = (int)(e / (NPAIR * NCL));
    const int* row = topk + (b * NCL + c) * KTOP;
    int si = row[pi[p]], dj = row[pj[p]];
    float nrm = rsqrtf(deg[b * NPTS + si] * deg[b * NPTS + dj]);
    atomicAdd(&agg[((i64)(b * NPTS + dj)) * DIM + d], hw[((i64)(b * NPTS + si)) * DIM + d] * nrm);
    atomicAdd(&agg[((i64)(b * NPTS + si)) * DIM + d], hw[((i64)(b * NPTS + dj)) * DIM + d] * nrm);
}

// ---------------------------------------------------------------------------
// LayerNorm kernels (160 threads = 5 waves per row)
// ---------------------------------------------------------------------------
static __device__ inline float ln_block(float x, float* partial, float* stats)
{
    int d = threadIdx.x, w = d >> 5, l = d & 31;
    float s = waveRedSum(x);
    if (l == 0) partial[w] = s;
    __syncthreads();
    if (d == 0) { float m = 0; for (int i = 0; i < 5; ++i) m += partial[i]; stats[0] = m / (float)DIM; }
    __syncthreads();
    float mu = stats[0];
    float c = x - mu;
    float s2 = waveRedSum(c * c);
    if (l == 0) partial[w] = s2;
    __syncthreads();
    if (d == 0) { float m = 0; for (int i = 0; i < 5; ++i) m += partial[i]; stats[1] = rsqrtf(m / (float)DIM + 1e-5f); }
    __syncthreads();
    return c * stats[1];
}

__global__ __launch_bounds__(DIM)
void ln_gcn_k(const float* __restrict__ agg, const float* __restrict__ hw,
              const float* __restrict__ hres, const float* __restrict__ deg,
              const float* __restrict__ gb, const float* __restrict__ g,
              const float* __restrict__ be, float* __restrict__ out)
{
    __shared__ float partial[5]; __shared__ float stats[2];
    int r = blockIdx.x, d = threadIdx.x;
    i64 base = (i64)r * DIM;
    float x = agg[base + d] + hw[base + d] / deg[r] + gb[d] + hres[base + d];
    float y = ln_block(x, partial, stats);
    out[base + d] = y * g[d] + be[d];
}

__global__ __launch_bounds__(DIM)
void ln_add_k(const float* __restrict__ a, const float* __restrict__ b2,
              const float* __restrict__ g, const float* __restrict__ be,
              float* __restrict__ out)
{
    __shared__ float partial[5]; __shared__ float stats[2];
    int r = blockIdx.x, d = threadIdx.x;
    i64 base = (i64)r * DIM;
    float x = a[base + d] + b2[base + d];
    float y = ln_block(x, partial, stats);
    out[base + d] = y * g[d] + be[d];
}

__global__ void add_k(const float* __restrict__ a, const float* __restrict__ b,
                      float* __restrict__ o, i64 n)
{
    i64 i = (i64)blockIdx.x * blockDim.x + threadIdx.x;
    if (i < n) o[i] = a[i] + b[i];
}

// row softmax over 1024 with scale
__global__ __launch_bounds__(256)
void softmax_k(float* __restrict__ S)
{
    __shared__ float part[8]; __shared__ float st;
    i64 row = blockIdx.x;
    float* p = S + row * (i64)NPTS;
    int t = threadIdx.x;
    const float scale = 0.1767766952966369f;   // 1/sqrt(32)

    float mx = -1e30f;
    for (int j = t; j < NPTS; j += 256) mx = fmaxf(mx, p[j] * scale);
    mx = waveRedMax(mx);
    if ((t & 31) == 0) part[t >> 5] = mx;
    __syncthreads();
    if (t == 0) { float m = part[0]; for (int i = 1; i < 8; ++i) m = fmaxf(m, part[i]); st = m; }
    __syncthreads();
    float rmax = st;
    __syncthreads();

    float sum = 0.f;
    for (int j = t; j < NPTS; j += 256) {
        float e = __expf(p[j] * scale - rmax);
        p[j] = e; sum += e;
    }
    sum = waveRedSum(sum);
    if ((t & 31) == 0) part[t >> 5] = sum;
    __syncthreads();
    if (t == 0) { float m = 0; for (int i = 0; i < 8; ++i) m += part[i]; st = m; }
    __syncthreads();
    float inv = 1.f / st;
    for (int j = t; j < NPTS; j += 256) p[j] *= inv;
}

// mean pool over N
__global__ void pool_k(const float* __restrict__ h, float* __restrict__ pooled)
{
    int i = blockIdx.x * blockDim.x + threadIdx.x;
    if (i >= NBATCH * DIM) return;
    int d = i % DIM, b = i / DIM;
    float a = 0.f;
    for (int n = 0; n < NPTS; ++n) a += h[((i64)(b * NPTS + n)) * DIM + d];
    pooled[i] = a * (1.0f / NPTS);
}

__global__ void head1_k(const float* __restrict__ pooled, const float* __restrict__ w,
                        const float* __restrict__ be, float* __restrict__ t1)
{
    int i = blockIdx.x * blockDim.x + threadIdx.x;
    if (i >= NBATCH * DIM) return;
    int d = i % DIM, b = i / DIM;
    float a = be[d];
    for (int k = 0; k < DIM; ++k) a += pooled[b * DIM + k] * w[k * DIM + d];
    t1[i] = fmaxf(a, 0.f);
}

__global__ void head2_k(const float* __restrict__ t1, const float* __restrict__ w,
                        const float* __restrict__ be, float* __restrict__ out)
{
    int i = blockIdx.x * blockDim.x + threadIdx.x;
    if (i >= NBATCH * OUTD) return;
    int o = i % OUTD, b = i / OUTD;
    float a = be[o];
    for (int k = 0; k < DIM; ++k) a += t1[b * DIM + k] * w[k * OUTD + o];
    out[i] = a;
}

// ---------------------------------------------------------------------------
// launch
// ---------------------------------------------------------------------------
extern "C" void kernel_launch(void* const* d_in, const int* in_sizes, int n_in,
                              void* d_out, int out_size, void* d_ws, size_t ws_size,
                              hipStream_t stream)
{
    (void)in_sizes; (void)n_in; (void)out_size; (void)ws_size;
    const float* x      = (const float*)d_in[0];
    const float* wemb   = (const float*)d_in[1];
    const float* bemb   = (const float*)d_in[2];
    const float* cen    = (const float*)d_in[3];
    const float* gcn_w  = (const float*)d_in[4];
    const float* gcn_b  = (const float*)d_in[5];
    const float* wq     = (const float*)d_in[6];
    const float* wk     = (const float*)d_in[7];
    const float* wv     = (const float*)d_in[8];
    const float* wo     = (const float*)d_in[9];
    const float* bq     = (const float*)d_in[10];
    const float* bk     = (const float*)d_in[11];
    const float* bv     = (const float*)d_in[12];
    const float* bo     = (const float*)d_in[13];
    const float* ln1g   = (const float*)d_in[14];
    const float* ln1b   = (const float*)d_in[15];
    const float* ln2g   = (const float*)d_in[16];
    const float* ln2b   = (const float*)d_in[17];
    const float* ln3g   = (const float*)d_in[18];
    const float* ln3b   = (const float*)d_in[19];
    const float* mw1    = (const float*)d_in[20];
    const float* mb1    = (const float*)d_in[21];
    const float* mw2    = (const float*)d_in[22];
    const float* mb2    = (const float*)d_in[23];
    const float* l1w    = (const float*)d_in[24];
    const float* l1b    = (const float*)d_in[25];
    const float* l2w    = (const float*)d_in[26];
    const float* l2b    = (const float*)d_in[27];
    float* out = (float*)d_out;

    const i64 NBND = (i64)NBATCH * NPTS * DIM;        // 655360
    float* W = (float*)d_ws;
    float* h     = W;            W += NBND;
    float* hloc  = W;            W += NBND;
    float* hatt  = W;            W += NBND;
    float* hw    = W;            W += NBND;   // also MLP2 output
    float* agg   = W;            W += NBND;   // also Wo output
    float* h2    = W;            W += NBND;
    float* Q     = W;            W += NBND;
    float* Kb    = W;            W += NBND;
    float* V     = W;            W += NBND;
    float* hattT = W;            W += NBND;
    float* mlp1  = W;            W += (i64)NBATCH * NPTS * DFF;          // 2621440
    float* S     = W;            W += (i64)NBATCH * NHEADS * NPTS * NPTS; // 20971520
    float* sims  = W;            W += (i64)NBATCH * NCL * NPTS;
    float* deg   = W;            W += NBATCH * NPTS;
    float* pooled= W;            W += NBATCH * DIM;
    float* t1    = W;            W += NBATCH * DIM;
    int* topk = (int*)W;
    int* pi   = topk + NBATCH * NCL * KTOP;
    int* pj   = pi + NPAIR;

    auto gemm = [&](const float* A, i64 sAr, i64 soA, i64 siA,
                    const float* B, i64 sBr, i64 sBc, i64 soB, i64 siB,
                    const float* bias, i64 soBi, i64 siBi,
                    float* C, i64 sCr, i64 soC, i64 siC,
                    int M, int N, int Kd, int batch, int nInner, int relu) {
        int tiles = (M / 16) * (N / 32);
        dim3 grid((tiles + 7) / 8, batch);
        gemm_wmma<<<grid, 256, 0, stream>>>(A, sAr, soA, siA,
                                            B, sBr, sBc, soB, siB,
                                            bias, soBi, siBi,
                                            C, sCr, soC, siC,
                                            M, N, Kd, nInner, relu);
    };

    // ---- embedding, hypergraph construction (once) ----
    {
        i64 n = NBND;
        embed_k<<<(int)((n + 255) / 256), 256, 0, stream>>>(x, wemb, bemb, h);
        int ns = NBATCH * NCL * NPTS;
        sims_k<<<(ns + 255) / 256, 256, 0, stream>>>(h, cen, sims);
        topk_k<<<NBATCH * NCL, 1024, 0, stream>>>(sims, topk);
        pairs_k<<<(NPAIR + 255) / 256, 256, 0, stream>>>(pi, pj);
        deginit_k<<<(NBATCH * NPTS + 255) / 256, 256, 0, stream>>>(deg);
        int ne = NBATCH * NCL * NPAIR;
        degadd_k<<<(ne + 255) / 256, 256, 0, stream>>>(topk, pi, pj, deg);
    }

    const i64 W2 = (i64)DIM * DIM;
    const i64 headSoA = (i64)NPTS * DIM;   // per-b stride within [4096,160] views
    const i64 Ssz = (i64)NPTS * NPTS;

    for (int l = 0; l < NLAY; ++l) {
        // ---- GCN branch ----
        gemm(h, DIM, 0, 0,  gcn_w + l * W2, DIM, 1, 0, 0,
             nullptr, 0, 0,  hw, DIM, 0, 0,
             NBATCH * NPTS, DIM, DIM, 1, 1, 0);
        zero_k<<<(int)((NBND + 255) / 256), 256, 0, stream>>>(agg, NBND);
        {
            i64 tot = (i64)NBATCH * NCL * NPAIR * DIM;
            gcnagg_k<<<(int)((tot + 255) / 256), 256, 0, stream>>>(topk, pi, pj, deg, hw, agg);
        }
        ln_gcn_k<<<NBATCH * NPTS, DIM, 0, stream>>>(agg, hw, h, deg,
                                                    gcn_b + l * DIM, ln1g + l * DIM, ln1b + l * DIM, hloc);

        // ---- attention branch ----
        gemm(h, DIM, 0, 0,  wq + l * W2, DIM, 1, 0, 0,  bq + l * DIM, 0, 0,
             Q, DIM, 0, 0,  NBATCH * NPTS, DIM, DIM, 1, 1, 0);
        gemm(h, DIM, 0, 0,  wk + l * W2, DIM, 1, 0, 0,  bk + l * DIM, 0, 0,
             Kb, DIM, 0, 0,  NBATCH * NPTS, DIM, DIM, 1, 1, 0);
        gemm(h, DIM, 0, 0,  wv + l * W2, DIM, 1, 0, 0,  bv + l * DIM, 0, 0,
             V, DIM, 0, 0,  NBATCH * NPTS, DIM, DIM, 1, 1, 0);

        // S = Q @ K^T per (b,head): batch=20, nInner=NHEADS; B is K-contiguous
        gemm(Q,  DIM, headSoA, DH,
             Kb, 1, DIM, headSoA, DH,
             nullptr, 0, 0,
             S, NPTS, (i64)NHEADS * Ssz, Ssz,
             NPTS, NPTS, DH, NBATCH * NHEADS, NHEADS, 0);
        softmax_k<<<NBATCH * NHEADS * NPTS, 256, 0, stream>>>(S);
        // hattT = att @ V per (b,head)
        gemm(S, NPTS, (i64)NHEADS * Ssz, Ssz,
             V, DIM, 1, headSoA, DH,
             nullptr, 0, 0,
             hattT, DIM, headSoA, DH,
             NPTS, DH, NPTS, NBATCH * NHEADS, NHEADS, 0);
        // output projection (reuse agg)
        gemm(hattT, DIM, 0, 0,  wo + l * W2, DIM, 1, 0, 0,  bo + l * DIM, 0, 0,
             agg, DIM, 0, 0,  NBATCH * NPTS, DIM, DIM, 1, 1, 0);
        ln_add_k<<<NBATCH * NPTS, DIM, 0, stream>>>(agg, h, ln2g + l * DIM, ln2b + l * DIM, hatt);

        // ---- combine + MLP ----
        add_k<<<(int)((NBND + 255) / 256), 256, 0, stream>>>(hloc, hatt, h2, NBND);
        gemm(h2, DIM, 0, 0,  mw1 + (i64)l * DIM * DFF, DFF, 1, 0, 0,  mb1 + l * DFF, 0, 0,
             mlp1, DFF, 0, 0,  NBATCH * NPTS, DFF, DIM, 1, 1, 1 /*relu*/);
        gemm(mlp1, DFF, 0, 0,  mw2 + (i64)l * DFF * DIM, DIM, 1, 0, 0,  mb2 + l * DIM, 0, 0,
             hw, DIM, 0, 0,  NBATCH * NPTS, DIM, DFF, 1, 1, 0);
        ln_add_k<<<NBATCH * NPTS, DIM, 0, stream>>>(h2, hw, ln3g + l * DIM, ln3b + l * DIM, h);
    }

    // ---- readout ----
    pool_k<<<(NBATCH * DIM + 255) / 256, 256, 0, stream>>>(h, pooled);
    head1_k<<<(NBATCH * DIM + 255) / 256, 256, 0, stream>>>(pooled, l1w, l1b, t1);
    head2_k<<<(NBATCH * OUTD + 255) / 256, 256, 0, stream>>>(t1, l2w, l2b, out);
}
